// Model_25254407700539
// MI455X (gfx1250) — matrix operand order repaired and checked by reading, not compiled, not run on hardware
//
#include <hip/hip_runtime.h>

// ---------------- WMMA types (CDNA5 / gfx1250, wave32) ----------------
typedef __attribute__((ext_vector_type(16))) __bf16 v16bf;
typedef __attribute__((ext_vector_type(8)))  float  v8f;

// H = hidden dim = 64 everywhere.
#define HDIM 64

// ---- Fragment loaders (layouts per CDNA5 ISA 7.12.2, wave32) ----
// A-matrix 16x32 bf16: lane L holds row M = L&15; element e holds
//   K = ktile*32 + 8*(L>=16) + (e<8 ? e : e+8)
__device__ __forceinline__ v16bf load_a_bf16(const float* __restrict__ A,
                                             int row, int hi, int ktile) {
  const float* ar = A + (size_t)row * HDIM + ktile * 32 + hi * 8;
  v16bf a;
#pragma unroll
  for (int e = 0; e < 8; ++e) a[e] = (__bf16)ar[e];
#pragma unroll
  for (int e = 8; e < 16; ++e) a[e] = (__bf16)ar[e + 8];
  return a;
}

// B-matrix 32x16 bf16 from row-major W[64][64]: lane gives N, element e gives
//   K = ktile*32 + 8*(lane>=16) + (e<8 ? e : e+8) ; column n passed in explicitly.
__device__ __forceinline__ v16bf load_b_bf16(const float* __restrict__ W,
                                             int n, int hi, int ktile) {
  const float* wr = W + (size_t)(ktile * 32 + hi * 8) * HDIM + n;
  v16bf b;
#pragma unroll
  for (int e = 0; e < 16; ++e) b[e] = (__bf16)wr[(size_t)(e < 8 ? e : e + 8) * HDIM];
  return b;
}

// ---------------- utility kernels ----------------
__global__ void fill_zero_kernel(float* __restrict__ p, size_t n) {
  size_t i = (size_t)blockIdx.x * blockDim.x + threadIdx.x;
  size_t st = (size_t)gridDim.x * blockDim.x;
  for (; i < n; i += st) p[i] = 0.0f;
}

__global__ void degree_kernel(const int* __restrict__ src,
                              const int* __restrict__ dst,
                              float* __restrict__ deg_p,
                              float* __restrict__ deg_t, int E) {
  int e = blockIdx.x * blockDim.x + threadIdx.x;
  if (e >= E) return;
  atomicAdd(&deg_t[dst[e]], 1.0f);
  atomicAdd(&deg_p[src[e]], 1.0f);
}

// x_out[i] = emb[nid[i]]  (row copy, float4 granularity: 16 qwords per row)
__global__ void gather_rows_kernel(const float* __restrict__ emb,
                                   const int* __restrict__ nid,
                                   float* __restrict__ xout, int N) {
  int t = blockIdx.x * blockDim.x + threadIdx.x;
  int i = t >> 4, q = t & 15;
  if (i >= N) return;
  const float4* s = (const float4*)(emb + (size_t)nid[i] * HDIM);
  float4* d = (float4*)(xout + (size_t)i * HDIM);
  d[q] = s[q];
}

// sum[dst[e]] += X[src[e]]  (64 threads per edge)
__global__ void scatter_add_kernel(const float* __restrict__ X,
                                   const int* __restrict__ srcIdx,
                                   const int* __restrict__ dstIdx,
                                   float* __restrict__ sum, int E) {
  size_t t = (size_t)blockIdx.x * blockDim.x + threadIdx.x;
  int e = (int)(t >> 6);
  int d = (int)(t & 63);
  if (e >= E) return;
  int s = srcIdx[e];
  int dd = dstIdx[e];
  atomicAdd(&sum[(size_t)dd * HDIM + d], X[(size_t)s * HDIM + d]);
}

// sum[i][d] /= max(deg[i],1)
__global__ void mean_div_kernel(float* __restrict__ sum,
                                const float* __restrict__ deg, size_t n64) {
  size_t t = (size_t)blockIdx.x * blockDim.x + threadIdx.x;
  if (t >= n64) return;
  sum[t] = sum[t] / fmaxf(deg[t >> 6], 1.0f);
}

// ---------------- WMMA GEMM kernels ----------------
// Per wave: one 16x16 tile of Y. Block = 256 thr = 8 waves = 32 rows x 64 cols.
// Y = opt_relu(meanA @ WL + bl + X @ WR)
__global__ __launch_bounds__(256) void sage_gemm_kernel(
    const float* __restrict__ meanA, const float* __restrict__ X,
    const float* __restrict__ WL, const float* __restrict__ WR,
    const float* __restrict__ bl, float* __restrict__ Y, int N, int do_relu) {
  int wave = threadIdx.x >> 5;   // 0..7
  int lane = threadIdx.x & 31;
  int hi = lane >> 4;            // half-wave select
  int lcol = lane & 15;
  int rowBlock = blockIdx.x * 2 + (wave >> 2);
  int ct = wave & 3;             // column tile 0..3
  int row0 = rowBlock * 16;
  if (row0 >= N) return;         // uniform per wave (N multiple of 32 here)
  int row = row0 + lcol;         // A-fragment row for this lane
  int col = ct * 16 + lcol;      // B/C column for this lane

  v8f c = {};
#pragma unroll
  for (int kt = 0; kt < 2; ++kt) {
    v16bf aL = load_a_bf16(meanA, row, hi, kt);
    v16bf bL = load_b_bf16(WL, col, hi, kt);
    c = __builtin_amdgcn_wmma_f32_16x16x32_bf16(false, aL, false, bL,
                                                (short)0, c, false, false);
    v16bf aR = load_a_bf16(X, row, hi, kt);
    v16bf bR = load_b_bf16(WR, col, hi, kt);
    c = __builtin_amdgcn_wmma_f32_16x16x32_bf16(false, aR, false, bR,
                                                (short)0, c, false, false);
  }

  float bias = bl[col];
#pragma unroll
  for (int r = 0; r < 8; ++r) {
    int m = row0 + r + 8 * hi;   // C/D layout: VGPR r -> M=r (lo) / 8+r (hi)
    float v = c[r] + bias;
    if (do_relu) v = fmaxf(v, 0.0f);
    Y[(size_t)m * HDIM + col] = v;
  }
}

// x_t = x_taxon @ W + b + taxon_emb[nid]
__global__ __launch_bounds__(256) void taxon_init_kernel(
    const float* __restrict__ Xt, const float* __restrict__ W,
    const float* __restrict__ b, const float* __restrict__ emb,
    const int* __restrict__ nid, float* __restrict__ Y, int N) {
  int wave = threadIdx.x >> 5;
  int lane = threadIdx.x & 31;
  int hi = lane >> 4;
  int lcol = lane & 15;
  int rowBlock = blockIdx.x * 2 + (wave >> 2);
  int ct = wave & 3;
  int row0 = rowBlock * 16;
  if (row0 >= N) return;
  int row = row0 + lcol;
  int col = ct * 16 + lcol;

  v8f c = {};
#pragma unroll
  for (int kt = 0; kt < 2; ++kt) {
    v16bf a = load_a_bf16(Xt, row, hi, kt);
    v16bf bb = load_b_bf16(W, col, hi, kt);
    c = __builtin_amdgcn_wmma_f32_16x16x32_bf16(false, a, false, bb,
                                                (short)0, c, false, false);
  }
  float bias = b[col];
#pragma unroll
  for (int r = 0; r < 8; ++r) {
    int m = row0 + r + 8 * hi;
    Y[(size_t)m * HDIM + col] =
        c[r] + bias + emb[(size_t)nid[m] * HDIM + col];
  }
}

// out[e] = dot(P[es[e]], T[ed[e]])  (16 lanes per edge, float4 loads)
__global__ __launch_bounds__(256) void edge_dot_kernel(
    const float* __restrict__ P, const float* __restrict__ T,
    const int* __restrict__ es, const int* __restrict__ ed,
    float* __restrict__ out, int EL) {
  int t = blockIdx.x * blockDim.x + threadIdx.x;
  int e = t >> 4;
  int l = t & 15;
  if (e >= EL) return;
  const float4* pr = (const float4*)(P + (size_t)es[e] * HDIM);
  const float4* tr = (const float4*)(T + (size_t)ed[e] * HDIM);
  float4 a = pr[l];
  float4 bv = tr[l];
  float v = a.x * bv.x + a.y * bv.y + a.z * bv.z + a.w * bv.w;
  v += __shfl_xor(v, 8, 32);
  v += __shfl_xor(v, 4, 32);
  v += __shfl_xor(v, 2, 32);
  v += __shfl_xor(v, 1, 32);
  if (l == 0) out[e] = v;
}

// ---------------- launch ----------------
extern "C" void kernel_launch(void* const* d_in, const int* in_sizes, int n_in,
                              void* d_out, int out_size, void* d_ws,
                              size_t ws_size, hipStream_t stream) {
  const float* x_taxon   = (const float*)d_in[0];
  const float* tax_lin_w = (const float*)d_in[1];
  const float* tax_lin_b = (const float*)d_in[2];
  const float* palm_emb  = (const float*)d_in[3];
  const float* taxon_emb = (const float*)d_in[4];
  const int*   n_id_p    = (const int*)d_in[5];
  const int*   n_id_t    = (const int*)d_in[6];
  const int*   esrc      = (const int*)d_in[7];
  const int*   edst      = (const int*)d_in[8];
  const int*   elsrc     = (const int*)d_in[9];
  const int*   eldst     = (const int*)d_in[10];
  const float* c1pt_wl = (const float*)d_in[11];
  const float* c1pt_wr = (const float*)d_in[12];
  const float* c1tp_wl = (const float*)d_in[13];
  const float* c1tp_wr = (const float*)d_in[14];
  const float* c2pt_wl = (const float*)d_in[15];
  const float* c2pt_wr = (const float*)d_in[16];
  const float* c2tp_wl = (const float*)d_in[17];
  const float* c2tp_wr = (const float*)d_in[18];
  const float* c1pt_bl = (const float*)d_in[19];
  const float* c1tp_bl = (const float*)d_in[20];
  const float* c2pt_bl = (const float*)d_in[21];
  const float* c2tp_bl = (const float*)d_in[22];

  const int NP = in_sizes[5];   // 200000
  const int NT = in_sizes[6];   // 100000
  const int E  = in_sizes[7];   // 2000000
  const int EL = in_sizes[9];   // 500000

  // workspace carve-out (floats); layer-2 outputs alias x_t / x_p
  float* ws   = (float*)d_ws;
  float* x_p  = ws; ws += (size_t)NP * HDIM;
  float* x_t  = ws; ws += (size_t)NT * HDIM;
  float* sum_t = ws; ws += (size_t)NT * HDIM;
  float* sum_p = ws; ws += (size_t)NP * HDIM;
  float* t1   = ws; ws += (size_t)NT * HDIM;
  float* p1   = ws; ws += (size_t)NP * HDIM;
  float* deg_t = ws; ws += NT;
  float* deg_p = ws; ws += NP;
  float* t2 = x_t;  // x_t dead after layer-1 GEMMs
  float* p2 = x_p;  // x_p dead after layer-1 GEMMs
  float* outp = (float*)d_out;

  const int TB = 256;
  dim3 blk(TB);
  size_t nt64 = (size_t)NT * HDIM, np64 = (size_t)NP * HDIM;
  unsigned scat_grid = (unsigned)(((size_t)E * HDIM + TB - 1) / TB);

  // ---- degrees (edge-list only; shared by both layers) ----
  fill_zero_kernel<<<1024, blk, 0, stream>>>(deg_t, (size_t)NT);
  fill_zero_kernel<<<1024, blk, 0, stream>>>(deg_p, (size_t)NP);
  degree_kernel<<<(E + TB - 1) / TB, blk, 0, stream>>>(esrc, edst, deg_p,
                                                       deg_t, E);

  // ---- initial features ----
  gather_rows_kernel<<<((size_t)NP * 16 + TB - 1) / TB, blk, 0, stream>>>(
      palm_emb, n_id_p, x_p, NP);
  taxon_init_kernel<<<(NT + 31) / 32, blk, 0, stream>>>(
      x_taxon, tax_lin_w, tax_lin_b, taxon_emb, n_id_t, x_t, NT);

  // ---- layer 1 ----
  fill_zero_kernel<<<2048, blk, 0, stream>>>(sum_t, nt64);
  fill_zero_kernel<<<2048, blk, 0, stream>>>(sum_p, np64);
  scatter_add_kernel<<<scat_grid, blk, 0, stream>>>(x_p, esrc, edst, sum_t, E);
  scatter_add_kernel<<<scat_grid, blk, 0, stream>>>(x_t, edst, esrc, sum_p, E);
  mean_div_kernel<<<(nt64 + TB - 1) / TB, blk, 0, stream>>>(sum_t, deg_t, nt64);
  mean_div_kernel<<<(np64 + TB - 1) / TB, blk, 0, stream>>>(sum_p, deg_p, np64);
  sage_gemm_kernel<<<(NT + 31) / 32, blk, 0, stream>>>(sum_t, x_t, c1pt_wl,
                                                       c1pt_wr, c1pt_bl, t1,
                                                       NT, 1);
  sage_gemm_kernel<<<(NP + 31) / 32, blk, 0, stream>>>(sum_p, x_p, c1tp_wl,
                                                       c1tp_wr, c1tp_bl, p1,
                                                       NP, 1);

  // ---- layer 2 ----
  fill_zero_kernel<<<2048, blk, 0, stream>>>(sum_t, nt64);
  fill_zero_kernel<<<2048, blk, 0, stream>>>(sum_p, np64);
  scatter_add_kernel<<<scat_grid, blk, 0, stream>>>(p1, esrc, edst, sum_t, E);
  scatter_add_kernel<<<scat_grid, blk, 0, stream>>>(t1, edst, esrc, sum_p, E);
  mean_div_kernel<<<(nt64 + TB - 1) / TB, blk, 0, stream>>>(sum_t, deg_t, nt64);
  mean_div_kernel<<<(np64 + TB - 1) / TB, blk, 0, stream>>>(sum_p, deg_p, np64);
  sage_gemm_kernel<<<(NT + 31) / 32, blk, 0, stream>>>(sum_t, t1, c2pt_wl,
                                                       c2pt_wr, c2pt_bl, t2,
                                                       NT, 0);
  sage_gemm_kernel<<<(NP + 31) / 32, blk, 0, stream>>>(sum_p, p1, c2tp_wl,
                                                       c2tp_wr, c2tp_bl, p2,
                                                       NP, 0);

  // ---- classifier ----
  edge_dot_kernel<<<((size_t)EL * 16 + TB - 1) / TB, blk, 0, stream>>>(
      p2, t2, elsrc, eldst, outp, EL);
}
